// SDPASparseBoolMask_36017595744258
// MI455X (gfx1250) — matrix operand order repaired and checked by reading, not compiled, not run on hardware
//
#include <hip/hip_runtime.h>

// ---------------------------------------------------------------------------
// Flash-attention (SDPA with first-half-KV boolean mask) for gfx1250 / MI455X
//   B=2, H=16, SQ=2048, KV=4096 (effective 2048), D=128, f32 in/out
//   Compute: v_wmma_f32_16x16x32_bf16; DPP16 row_ror reductions;
//   softmax in log2 domain (raw v_exp_f32), global_prefetch_b8 staging hints
// ---------------------------------------------------------------------------

typedef __attribute__((ext_vector_type(16))) __bf16 v16bf;
typedef __attribute__((ext_vector_type(8)))  float  v8f;

#define SQ_     2048
#define KV_     4096
#define KV_EFF  2048
#define D_      128
#define BH_     32      // B*H
#define QTILE   64      // q rows per block (4 waves x 16 rows)
#define KTILE   64      // kv positions staged per iteration
#define NWAVE   4

// padded LDS pitches (uints / halfwords): 16B alignment + conflict-free b128 reads
#define KT_P 68         // kt[kv][d/2]  row pitch (64 used)
#define VT_P 36         // vt[d][kv/2]  row pitch (32 used)
#define PB_P 40         // pb[row][n]   row pitch in halfwords (32 used)

struct U8x4 { uint4 a, b; };

static __device__ __forceinline__ unsigned pk2(float x, float y) {
    union { __bf16 h[2]; unsigned u; } r;
    r.h[0] = (__bf16)x; r.h[1] = (__bf16)y;
    return r.u;
}
static __device__ __forceinline__ unsigned short bfb(float x) {
    union { __bf16 h; unsigned short u; } r;
    r.h = (__bf16)x;
    return r.u;
}
static __device__ __forceinline__ v16bf lds_frag(const unsigned* p) {
    const uint4* p4 = (const uint4*)p;
    U8x4 u; u.a = p4[0]; u.b = p4[1];
    return __builtin_bit_cast(v16bf, u);
}
static __device__ __forceinline__ v8f wmma_bf16(v16bf a, v16bf b, v8f c) {
    return __builtin_amdgcn_wmma_f32_16x16x32_bf16(false, a, false, b, (short)0, c,
                                                   false, false);
}

// DPP16 row_ror:N move (stays within each 16-lane row of a wave32)
template <int CTRL>
static __device__ __forceinline__ float dpp_rot(float x) {
    return __builtin_bit_cast(
        float, __builtin_amdgcn_update_dpp(0, __builtin_bit_cast(int, x),
                                           CTRL, 0xf, 0xf, true));
}
// full 16-lane reduction via rotations (equivalent to xor tree over the group)
static __device__ __forceinline__ float red16_max(float x) {
    x = fmaxf(x, dpp_rot<0x121>(x));   // row_ror:1
    x = fmaxf(x, dpp_rot<0x122>(x));   // row_ror:2
    x = fmaxf(x, dpp_rot<0x124>(x));   // row_ror:4
    x = fmaxf(x, dpp_rot<0x128>(x));   // row_ror:8
    return x;
}
static __device__ __forceinline__ float red16_sum(float x) {
    x += dpp_rot<0x121>(x);
    x += dpp_rot<0x122>(x);
    x += dpp_rot<0x124>(x);
    x += dpp_rot<0x128>(x);
    return x;
}

__global__ __launch_bounds__(128)
void fa2_gfx1250_kernel(const float* __restrict__ q,
                        const float* __restrict__ k,
                        const float* __restrict__ v,
                        float* __restrict__ out)
{
    __shared__ unsigned       kt[KTILE][KT_P];        // packed bf16 d-pairs of K
    __shared__ unsigned       vt[D_][VT_P];           // packed bf16 kv-pairs of V
    __shared__ unsigned short pb[NWAVE][16][PB_P];    // per-wave P bounce buffer

    const int bh   = blockIdx.y;
    const int q0   = blockIdx.x * QTILE;
    const int tid  = threadIdx.x;
    const int lane = tid & 31;
    const int wave = tid >> 5;
    const int nn   = lane & 15;      // column index inside a 16-wide tile
    const int hi   = lane >> 4;      // lane-half select
    const int kbA  = hi ? 8 : 0;     // unused: A fragments also use kbB
    const int kbB  = hi ? 16 : 0;    // B-fragment K group offset
    // fold 1/sqrt(D) * log2(e) into Q so scores live in the log2 domain:
    //   softmax(x) = exp2(s - max(s)) / sum(...) with s = x * log2(e)/sqrt(D)
    const float scale = 0.08838834764831845f * 1.4426950408889634f;

    const float* qb = q + (size_t)bh * SQ_ * D_;
    const float* kb = k + (size_t)bh * KV_ * D_;
    const float* vb = v + (size_t)bh * KV_ * D_;
    float*       ob = out + (size_t)bh * SQ_ * D_;

    // ---- Q fragments: 16x128 rows for this wave, pre-scaled, A-layout ----
    v16bf qf[4];
    {
        const float* qr = qb + (size_t)(q0 + wave * 16 + nn) * D_;
#pragma unroll
        for (int s = 0; s < 4; ++s) {
#pragma unroll
            for (int j = 0; j < 8; ++j) {
                const int kk = s * 32 + kbB + 2 * j;
                const float2 x = *(const float2*)(qr + kk);
                qf[s][2 * j]     = (__bf16)(x.x * scale);
                qf[s][2 * j + 1] = (__bf16)(x.y * scale);
            }
        }
    }

    // ---- accumulators: O (16x128 f32 across wave), row max / row sum ----
    v8f o[8];
#pragma unroll
    for (int t = 0; t < 8; ++t) o[t] = (v8f){0.f, 0.f, 0.f, 0.f, 0.f, 0.f, 0.f, 0.f};
    float mrow[8], lrow[8];
#pragma unroll
    for (int r = 0; r < 8; ++r) { mrow[r] = -1e30f; lrow[r] = 0.f; }

    for (int kv0 = 0; kv0 < KV_EFF; kv0 += KTILE) {
        __syncthreads();  // previous-stage LDS reads complete before overwrite

        // stage K: kt[n][dp] = pack(K[kv0+n][2dp], K[kv0+n][2dp+1])
#pragma unroll 4
        for (int i = 0; i < (KTILE * (D_ / 2)) / 128; ++i) {
            const int idx = i * 128 + tid;
            const int n = idx >> 6, dp = idx & 63;
            const float2 x = *(const float2*)(kb + (size_t)(kv0 + n) * D_ + 2 * dp);
            kt[n][dp] = pk2(x.x, x.y);
        }
        // stage V: vt[d][kp] = pack(V[kv0+2kp][d], V[kv0+2kp+1][d])
#pragma unroll 4
        for (int i = 0; i < ((KTILE / 2) * D_) / 128; ++i) {
            const int idx = i * 128 + tid;
            const int d = idx & 127, kp = idx >> 7;
            const float a0 = vb[(size_t)(kv0 + 2 * kp) * D_ + d];
            const float a1 = vb[(size_t)(kv0 + 2 * kp + 1) * D_ + d];
            vt[d][kp] = pk2(a0, a1);
        }
        __syncthreads();

        // prefetch next stage (global_prefetch_b8)
        if (kv0 + KTILE < KV_EFF) {
            __builtin_prefetch(kb + (size_t)(kv0 + KTILE) * D_ + tid * 64, 0, 1);
            __builtin_prefetch(vb + (size_t)(kv0 + KTILE) * D_ + tid * 64, 0, 1);
        }

#pragma unroll
        for (int sub = 0; sub < 2; ++sub) {
            const int kvo = sub * 32;

            // ---- S = (Q*scale) @ K^T : two 16x16 f32 tiles, K-dim = 128 ----
            v8f s0 = (v8f){0.f,0.f,0.f,0.f,0.f,0.f,0.f,0.f};
            v8f s1 = s0;
#pragma unroll
            for (int st = 0; st < 4; ++st) {
                const int dp0 = st * 16 + (kbB >> 1);
                const v16bf b0 = lds_frag(&kt[kvo + nn][dp0]);
                const v16bf b1 = lds_frag(&kt[kvo + 16 + nn][dp0]);
                s0 = wmma_bf16(qf[st], b0, s0);
                s1 = wmma_bf16(qf[st], b1, s1);
            }

            // ---- online softmax (log2 domain): max / rescale / exp2 / sum ----
#pragma unroll
            for (int r = 0; r < 8; ++r) {
                const float mx    = red16_max(fmaxf(s0[r], s1[r]));
                const float mnew  = fmaxf(mrow[r], mx);
                const float alpha = __builtin_amdgcn_exp2f(mrow[r] - mnew);
                mrow[r] = mnew;
                const float e0 = __builtin_amdgcn_exp2f(s0[r] - mnew);
                const float e1 = __builtin_amdgcn_exp2f(s1[r] - mnew);
                s0[r] = e0; s1[r] = e1;
                lrow[r] = lrow[r] * alpha + red16_sum(e0 + e1);
#pragma unroll
                for (int t = 0; t < 8; ++t) o[t][r] *= alpha;
            }

            // ---- P: C-layout -> A-layout via per-wave LDS bounce ----
#pragma unroll
            for (int r = 0; r < 8; ++r) {
                const int row = r + 8 * hi;
                pb[wave][row][nn]      = bfb(s0[r]);
                pb[wave][row][16 + nn] = bfb(s1[r]);
            }
            // same-wave DS store->load ordering; also a compiler memory barrier
            asm volatile("s_wait_dscnt 0" ::: "memory");

            v16bf pA;
            {
                const unsigned short* prow = &pb[wave][nn][0];
                U8x4 u;
                u.a = *(const uint4*)(prow + kbB);        // pairs kbB/2 .. +3
                u.b = *(const uint4*)(prow + kbB + 8);    // pairs kbB/2+4 .. +7
                pA = __builtin_bit_cast(v16bf, u);
            }

            // ---- O += P @ V : eight 16x16 tiles over D, K-dim = 32 ----
#pragma unroll
            for (int t = 0; t < 8; ++t) {
                const int kp0 = (kvo + kbB) >> 1;
                const v16bf bv = lds_frag(&vt[t * 16 + nn][kp0]);
                o[t] = wmma_bf16(pA, bv, o[t]);
            }
        }
    }

    // ---- epilogue: normalize by row sum and store ----
#pragma unroll
    for (int r = 0; r < 8; ++r) {
        const int row = q0 + wave * 16 + r + 8 * hi;
        const float inv = 1.0f / lrow[r];
#pragma unroll
        for (int t = 0; t < 8; ++t) {
            ob[(size_t)row * D_ + t * 16 + nn] = o[t][r] * inv;
        }
    }
}

extern "C" void kernel_launch(void* const* d_in, const int* in_sizes, int n_in,
                              void* d_out, int out_size, void* d_ws, size_t ws_size,
                              hipStream_t stream) {
    const float* q = (const float*)d_in[0];
    const float* k = (const float*)d_in[1];
    const float* v = (const float*)d_in[2];
    float* out = (float*)d_out;
    (void)in_sizes; (void)n_in; (void)out_size; (void)d_ws; (void)ws_size;

    dim3 grid(SQ_ / QTILE, BH_);   // (32, 32)
    dim3 block(128);               // 4 waves of 32
    fa2_gfx1250_kernel<<<grid, block, 0, stream>>>(q, k, v, out);
}
